// Transformer_56401510531520
// MI455X (gfx1250) — compile-verified, matrix operand order checked by reading, not run on hardware
//
#include <hip/hip_runtime.h>
#include <hip/hip_bf16.h>

// ---------------------------------------------------------------------------
// GPT-2 forward for MI455X (gfx1250): bf16 WMMA everywhere, f32 accumulate.
// Weights are converted fp32->bf16 AND transposed to [N][K] once per call so
// the GEMM hot loop is pure 16B vector traffic. GEMM/attention tile fills use
// the gfx1250 async Global->LDS path (ASYNCcnt) when the builtins exist.
// ---------------------------------------------------------------------------

typedef __attribute__((ext_vector_type(16))) __bf16 v16bf;
typedef __attribute__((ext_vector_type(8)))  __bf16 v8bf;
typedef __attribute__((ext_vector_type(8)))  float  v8f;
typedef __attribute__((ext_vector_type(4)))  int    v4i;

#define AS1 __attribute__((address_space(1)))
#define AS3 __attribute__((address_space(3)))
typedef AS1 v4i* gv4i_p;   // global int32x4 pointer (async src)
typedef AS3 v4i* lv4i_p;   // LDS int32x4 pointer (async dst)

#if defined(__has_builtin)
#if __has_builtin(__builtin_amdgcn_global_load_async_to_lds_b128) && \
    __has_builtin(__builtin_amdgcn_s_wait_asynccnt)
#define USE_ASYNC_LDS 1
#endif
#endif

static constexpr int NB  = 2;
static constexpr int NT  = 2048;
static constexpr int ND  = 1024;
static constexpr int NH  = 16;
static constexpr int NHD = 64;
static constexpr int NL  = 12;
static constexpr int NF  = 4096;
static constexpr int NV  = 50257;
static constexpr int NVP = 50304;        // 393 * 128
static constexpr int NM  = NB * NT;      // 4096 rows

__device__ __forceinline__ __bf16 f2bf(float f) {
  unsigned u = __builtin_bit_cast(unsigned, f);
  unsigned r = (u + 0x7FFFu + ((u >> 16) & 1u)) >> 16;
  unsigned short s = (unsigned short)r;
  return __builtin_bit_cast(__bf16, s);
}

__device__ __forceinline__ v16bf combine8(v8bf lo, v8bf hi) {
  v16bf r;
#pragma unroll
  for (int j = 0; j < 8; ++j) { r[j] = lo[j]; r[j + 8] = hi[j]; }
  return r;
}

__device__ __forceinline__ v8f wmma_bf16(v16bf a, v16bf b, v8f c) {
  return __builtin_amdgcn_wmma_f32_16x16x32_bf16(false, a, false, b, (short)0, c,
                                                 false, false);
}

// ---------------------------------------------------------------------------
// fp32 [K,Nsrc] -> bf16 transposed [Npad,K] (per layer), zero-padding columns.
// dst[l][n][k] = src[l][k][n]  (n >= Nsrc -> 0)
// ---------------------------------------------------------------------------
__global__ void cvt_t_kernel(const float* __restrict__ src,
                             __bf16* __restrict__ dst, int layers, int K,
                             int Nsrc, int Npad) {
  const size_t per = (size_t)Npad * K;
  const size_t total = (size_t)layers * per;
  size_t i = (size_t)blockIdx.x * blockDim.x + threadIdx.x;
  const size_t stride = (size_t)gridDim.x * blockDim.x;
  for (; i < total; i += stride) {
    const int l = (int)(i / per);
    const size_t r = i % per;
    const int n = (int)(r / K);
    const int k = (int)(r % K);
    float v = 0.f;
    if (n < Nsrc) v = src[(size_t)l * K * Nsrc + (size_t)k * Nsrc + n];
    dst[i] = f2bf(v);
  }
}

// ---------------------------------------------------------------------------
// token + positional embedding: h[row, :] = tok_emb[x[row]] + pos_emb[row%T]
// ---------------------------------------------------------------------------
__global__ __launch_bounds__(256) void embed_kernel(
    const int* __restrict__ x, const float* __restrict__ tok,
    const float* __restrict__ pos, float* __restrict__ h) {
  const int row = blockIdx.x;          // b*T + t
  const int t   = row % NT;
  const int id  = x[row];
#pragma unroll
  for (int i = 0; i < 4; ++i) {
    int d = threadIdx.x + i * 256;
    h[(size_t)row * ND + d] = tok[(size_t)id * ND + d] + pos[(size_t)t * ND + d];
  }
}

// ---------------------------------------------------------------------------
// LayerNorm over D=1024; emits f32 (residual source) and bf16 (GEMM A input)
// ---------------------------------------------------------------------------
__global__ __launch_bounds__(256) void ln_kernel(
    const float* __restrict__ X, const float* __restrict__ g,
    const float* __restrict__ bta, float* __restrict__ Yf,
    __bf16* __restrict__ Yb) {
  const int row = blockIdx.x;
  const float* x = X + (size_t)row * ND;
  float v4[4];
  float s = 0.f, sq = 0.f;
#pragma unroll
  for (int i = 0; i < 4; ++i) {
    float val = x[threadIdx.x + i * 256];
    v4[i] = val;
    s += val;
    sq += val * val;
  }
#pragma unroll
  for (int off = 16; off >= 1; off >>= 1) {
    s  += __shfl_xor(s, off, 32);
    sq += __shfl_xor(sq, off, 32);
  }
  __shared__ float ss[8], ssq[8];
  const int wave = threadIdx.x >> 5, lane = threadIdx.x & 31;
  if (lane == 0) { ss[wave] = s; ssq[wave] = sq; }
  __syncthreads();
  float total = 0.f, totsq = 0.f;
#pragma unroll
  for (int w = 0; w < 8; ++w) { total += ss[w]; totsq += ssq[w]; }
  const float mean = total * (1.f / ND);
  const float var  = totsq * (1.f / ND) - mean * mean;
  const float rstd = rsqrtf(var + 1e-5f);
#pragma unroll
  for (int i = 0; i < 4; ++i) {
    int d = threadIdx.x + i * 256;
    float yv = (v4[i] - mean) * rstd * g[d] + bta[d];
    Yf[(size_t)row * ND + d] = yv;
    Yb[(size_t)row * ND + d] = f2bf(yv);
  }
}

// ---------------------------------------------------------------------------
// Tiled bf16 GEMM: C[M,Nout] = A[M,K] x Bt[Ndim,K]^T + bias (+resid, relu)
// Block tile 128x128, K-step 32, 8 waves (4m x 2n), wave tile 32x64 ->
// acc[2][4], 8 WMMA per K-step. Bt is pre-transposed [n][k] so both tiles
// load/store as aligned 16B vectors; only the epilogue guards n < Nout.
// Tile fill uses global_load_async_to_lds_b128 (ASYNCcnt) when available.
// ---------------------------------------------------------------------------
__global__ __launch_bounds__(256) void gemm_bf16_kernel(
    const __bf16* __restrict__ A, const __bf16* __restrict__ Bt,
    const float* __restrict__ bias, const float* __restrict__ resid,
    float* __restrict__ Cf, __bf16* __restrict__ Cb, int Ndim, int Kdim,
    int Nout, int relu) {
  __shared__ __align__(16) __bf16 As[128][40];
  __shared__ __align__(16) __bf16 Bs[128][40];

  const int tid  = threadIdx.x;
  const int lane = tid & 31, wave = tid >> 5;
  const int wm = wave >> 1, wn = wave & 1;     // 4 x 2 wave grid
  const int m0 = blockIdx.y * 128;
  const int n0 = blockIdx.x * 128;
  const int lh = lane & 15, half = lane >> 4, h0 = half * 8;

  v8f acc[2][4] = {};

  const int lrow = tid >> 1;          // 0..127
  const int lkc  = (tid & 1) * 16;    // 0 or 16

  for (int k0 = 0; k0 < Kdim; k0 += 32) {
    const __bf16* ap = A + (size_t)(m0 + lrow) * Kdim + k0 + lkc;
    const __bf16* bp = Bt + (size_t)(n0 + lrow) * Kdim + k0 + lkc;
#ifdef USE_ASYNC_LDS
    __builtin_amdgcn_global_load_async_to_lds_b128(
        (gv4i_p)ap, (lv4i_p)&As[lrow][lkc], 0, 0);
    __builtin_amdgcn_global_load_async_to_lds_b128(
        (gv4i_p)(ap + 8), (lv4i_p)&As[lrow][lkc + 8], 0, 0);
    __builtin_amdgcn_global_load_async_to_lds_b128(
        (gv4i_p)bp, (lv4i_p)&Bs[lrow][lkc], 0, 0);
    __builtin_amdgcn_global_load_async_to_lds_b128(
        (gv4i_p)(bp + 8), (lv4i_p)&Bs[lrow][lkc + 8], 0, 0);
    __builtin_amdgcn_s_wait_asynccnt(0);
#else
    *(v8bf*)&As[lrow][lkc]     = *(const v8bf*)ap;
    *(v8bf*)&As[lrow][lkc + 8] = *(const v8bf*)(ap + 8);
    *(v8bf*)&Bs[lrow][lkc]     = *(const v8bf*)bp;
    *(v8bf*)&Bs[lrow][lkc + 8] = *(const v8bf*)(bp + 8);
#endif
    __syncthreads();

    v16bf af[2], bfr[4];
#pragma unroll
    for (int mt = 0; mt < 2; ++mt) {
      const int row = wm * 32 + mt * 16 + lh;
      af[mt] = combine8(*(const v8bf*)&As[row][h0],
                        *(const v8bf*)&As[row][h0 + 16]);
    }
#pragma unroll
    for (int nt = 0; nt < 4; ++nt) {
      const int col = wn * 64 + nt * 16 + lh;
      bfr[nt] = combine8(*(const v8bf*)&Bs[col][h0],
                         *(const v8bf*)&Bs[col][h0 + 16]);
    }
#pragma unroll
    for (int mt = 0; mt < 2; ++mt)
#pragma unroll
      for (int nt = 0; nt < 4; ++nt)
        acc[mt][nt] = wmma_bf16(af[mt], bfr[nt], acc[mt][nt]);
    __syncthreads();
  }

  // Epilogue: C layout -> lane holds column (lh), VGPR v holds row v+8*half
#pragma unroll
  for (int mt = 0; mt < 2; ++mt) {
#pragma unroll
    for (int nt = 0; nt < 4; ++nt) {
      const int n = n0 + wn * 64 + nt * 16 + lh;
      if (n < Nout) {
        const float bv = bias[n];
#pragma unroll
        for (int v = 0; v < 8; ++v) {
          const int m = m0 + wm * 32 + mt * 16 + v + half * 8;
          float xv = acc[mt][nt][v] + bv;
          if (relu) xv = fmaxf(xv, 0.f);
          if (resid) xv += resid[(size_t)m * Nout + n];
          if (Cf) Cf[(size_t)m * Nout + n] = xv;
          if (Cb) Cb[(size_t)m * Nout + n] = f2bf(xv);
        }
      }
    }
  }
  (void)Ndim;
}

// ---------------------------------------------------------------------------
// Flash attention: one block per (64-query tile, batch*head). 4 waves,
// each owns 16 query rows. QK^T and P·V via WMMA, online softmax with
// butterfly __shfl_xor reductions (rows 0-7 live in lanes 0-15, 8-15 in
// lanes 16-31, so xor masks 1,2,4,8 reduce across the 16 key columns).
// ---------------------------------------------------------------------------
__global__ __launch_bounds__(128) void attn_kernel(
    const __bf16* __restrict__ Q, const __bf16* __restrict__ Kb,
    const __bf16* __restrict__ Vb, __bf16* __restrict__ O) {
  const int qt   = blockIdx.x;         // query tile of 64
  const int bh   = blockIdx.y;         // b*H + h
  const int b    = bh / NH, hh = bh % NH;
  const int lane = threadIdx.x & 31;
  const int wave = threadIdx.x >> 5;   // 0..3
  const int qrow0 = qt * 64 + wave * 16;
  const int lh = lane & 15, half = lane >> 4, h0 = half * 8;

  __shared__ __align__(16) __bf16 Ks[32][NHD + 8];  // [key][hd]
  __shared__ __align__(16) __bf16 Vt[NHD][32 + 8];  // [hd][key]
  __shared__ __align__(16) __bf16 Ps[4][16][32 + 8];

  // Load Q fragments: 16x64 split into two 16x32 A fragments
  v16bf qf[2];
  {
    const int m = qrow0 + lh;
    const __bf16* qp = Q + ((size_t)(b * NT + m)) * ND + hh * NHD;
#pragma unroll
    for (int c = 0; c < 2; ++c)
      qf[c] = combine8(*(const v8bf*)(qp + c * 32 + h0),
                       *(const v8bf*)(qp + c * 32 + h0 + 16));
  }

  v8f oacc[4] = {};
  float mrun[8], lsum[8];
#pragma unroll
  for (int v = 0; v < 8; ++v) { mrun[v] = -1e30f; lsum[v] = 0.f; }

  const int nkb = 2 * qt + 2;          // causal: key blocks of 32
  for (int kb = 0; kb < nkb; ++kb) {
    {  // cooperative K/V block load; V stored transposed
      const int i = threadIdx.x;       // 0..127
      const int key = i >> 2;          // 0..31
      const int hdb = (i & 3) * 16;
      const size_t base =
          ((size_t)(b * NT + kb * 32 + key)) * ND + hh * NHD + hdb;
#ifdef USE_ASYNC_LDS
      __builtin_amdgcn_global_load_async_to_lds_b128(
          (gv4i_p)(Kb + base), (lv4i_p)&Ks[key][hdb], 0, 0);
      __builtin_amdgcn_global_load_async_to_lds_b128(
          (gv4i_p)(Kb + base + 8), (lv4i_p)&Ks[key][hdb + 8], 0, 0);
#else
      *(v8bf*)&Ks[key][hdb]     = *(const v8bf*)(Kb + base);
      *(v8bf*)&Ks[key][hdb + 8] = *(const v8bf*)(Kb + base + 8);
#endif
      v8bf v0 = *(const v8bf*)(Vb + base);
      v8bf v1 = *(const v8bf*)(Vb + base + 8);
#pragma unroll
      for (int j = 0; j < 8; ++j) {
        Vt[hdb + j][key]     = v0[j];
        Vt[hdb + 8 + j][key] = v1[j];
      }
#ifdef USE_ASYNC_LDS
      __builtin_amdgcn_s_wait_asynccnt(0);
#endif
    }
    __syncthreads();

    // S = Q K^T : two 16x16 score tiles (keys 0-15, 16-31 of block)
    v8f s[2];
#pragma unroll
    for (int nt = 0; nt < 2; ++nt) {
      v8f a = {};
      const int key = nt * 16 + lh;
#pragma unroll
      for (int c = 0; c < 2; ++c) {
        v16bf kf = combine8(*(const v8bf*)&Ks[key][c * 32 + h0],
                            *(const v8bf*)&Ks[key][c * 32 + h0 + 16]);
        a = wmma_bf16(qf[c], kf, a);
      }
      s[nt] = a;
    }

    // scale + causal mask
#pragma unroll
    for (int nt = 0; nt < 2; ++nt) {
      const int kg = kb * 32 + nt * 16 + lh;
#pragma unroll
      for (int v = 0; v < 8; ++v) {
        const int qg = qrow0 + v + half * 8;
        const float val = s[nt][v] * 0.125f;  // HD^-0.5
        s[nt][v] = (kg <= qg) ? val : -1e30f;
      }
    }

    // online softmax update
    float mnew[8], alpha[8];
#pragma unroll
    for (int v = 0; v < 8; ++v) {
      float mb = fmaxf(s[0][v], s[1][v]);
#pragma unroll
      for (int off = 8; off >= 1; off >>= 1)
        mb = fmaxf(mb, __shfl_xor(mb, off, 32));
      mnew[v]  = fmaxf(mrun[v], mb);
      alpha[v] = __expf(mrun[v] - mnew[v]);
      mrun[v]  = mnew[v];
      lsum[v] *= alpha[v];
    }
#pragma unroll
    for (int nt = 0; nt < 4; ++nt)
#pragma unroll
      for (int v = 0; v < 8; ++v) oacc[nt][v] *= alpha[v];

    // P = exp(S - m); stage into LDS to convert C-layout -> A-layout
#pragma unroll
    for (int nt = 0; nt < 2; ++nt)
#pragma unroll
      for (int v = 0; v < 8; ++v) {
        const float p = __expf(s[nt][v] - mnew[v]);
        s[nt][v] = p;
        Ps[wave][v + half * 8][nt * 16 + lh] = f2bf(p);
      }
#pragma unroll
    for (int v = 0; v < 8; ++v) {
      float rs = s[0][v] + s[1][v];
#pragma unroll
      for (int off = 8; off >= 1; off >>= 1) rs += __shfl_xor(rs, off, 32);
      lsum[v] += rs;
    }

    v16bf pf = combine8(*(const v8bf*)&Ps[wave][lh][h0],
                        *(const v8bf*)&Ps[wave][lh][h0 + 16]);
    // O += P V : 4 HD tiles of 16
#pragma unroll
    for (int nt = 0; nt < 4; ++nt) {
      const int hd = nt * 16 + lh;
      v16bf vf = combine8(*(const v8bf*)&Vt[hd][h0],
                          *(const v8bf*)&Vt[hd][h0 + 16]);
      oacc[nt] = wmma_bf16(pf, vf, oacc[nt]);
    }
    __syncthreads();
  }

  // normalize + store bf16 output at d = h*HD + hd
#pragma unroll
  for (int nt = 0; nt < 4; ++nt)
#pragma unroll
    for (int v = 0; v < 8; ++v) {
      const int t = qrow0 + v + half * 8;
      const float ov = oacc[nt][v] / lsum[v];
      O[((size_t)(b * NT + t)) * ND + hh * NHD + nt * 16 + lh] = f2bf(ov);
    }
}

// ---------------------------------------------------------------------------
// Host orchestration
// ---------------------------------------------------------------------------
extern "C" void kernel_launch(void* const* d_in, const int* in_sizes, int n_in,
                              void* d_out, int out_size, void* d_ws,
                              size_t ws_size, hipStream_t stream) {
  (void)in_sizes; (void)n_in; (void)out_size; (void)ws_size;

  const int*   x     = (const int*)d_in[0];
  const float* tok   = (const float*)d_in[1];
  const float* pos   = (const float*)d_in[2];
  const float* ln1g  = (const float*)d_in[3];
  const float* ln1b  = (const float*)d_in[4];
  const float* wq    = (const float*)d_in[5];
  const float* bq    = (const float*)d_in[6];
  const float* wk    = (const float*)d_in[7];
  const float* bk    = (const float*)d_in[8];
  const float* wv    = (const float*)d_in[9];
  const float* bv    = (const float*)d_in[10];
  const float* wo    = (const float*)d_in[11];
  const float* bo    = (const float*)d_in[12];
  const float* ln2g  = (const float*)d_in[13];
  const float* ln2b  = (const float*)d_in[14];
  const float* w1    = (const float*)d_in[15];
  const float* b1    = (const float*)d_in[16];
  const float* w2    = (const float*)d_in[17];
  const float* b2    = (const float*)d_in[18];
  const float* lnfg  = (const float*)d_in[19];
  const float* lnfb  = (const float*)d_in[20];
  const float* fcw   = (const float*)d_in[21];
  const float* fcb   = (const float*)d_in[22];
  float* logits = (float*)d_out;

  char* ws = (char*)d_ws;
  size_t off = 0;
  auto take = [&](size_t elems, size_t esz) -> void* {
    void* p = ws + off;
    off += (elems * esz + 255) & ~(size_t)255;
    return p;
  };

  const size_t LDD = (size_t)NL * ND * ND;
  const size_t LDF = (size_t)NL * ND * NF;
  __bf16* wq_t  = (__bf16*)take(LDD, 2);               // [L][N=D][K=D]
  __bf16* wk_t  = (__bf16*)take(LDD, 2);
  __bf16* wv_t  = (__bf16*)take(LDD, 2);
  __bf16* wo_t  = (__bf16*)take(LDD, 2);
  __bf16* w1_t  = (__bf16*)take(LDF, 2);               // [L][N=F][K=D]
  __bf16* w2_t  = (__bf16*)take(LDF, 2);               // [L][N=D][K=F]
  __bf16* fc_t  = (__bf16*)take((size_t)NVP * ND, 2);  // [N=50304][K=D]
  float*  h     = (float*)take((size_t)NM * ND, 4);
  float*  hnf   = (float*)take((size_t)NM * ND, 4);
  __bf16* hnb   = (__bf16*)take((size_t)NM * ND, 2);
  __bf16* qb    = (__bf16*)take((size_t)NM * ND, 2);
  __bf16* kb    = (__bf16*)take((size_t)NM * ND, 2);
  __bf16* vb    = (__bf16*)take((size_t)NM * ND, 2);
  __bf16* ab    = (__bf16*)take((size_t)NM * ND, 2);
  __bf16* ffnb  = (__bf16*)take((size_t)NM * NF, 2);

  const dim3 cvg(8192), cvb(256);
  cvt_t_kernel<<<cvg, cvb, 0, stream>>>(wq, wq_t, NL, ND, ND, ND);
  cvt_t_kernel<<<cvg, cvb, 0, stream>>>(wk, wk_t, NL, ND, ND, ND);
  cvt_t_kernel<<<cvg, cvb, 0, stream>>>(wv, wv_t, NL, ND, ND, ND);
  cvt_t_kernel<<<cvg, cvb, 0, stream>>>(wo, wo_t, NL, ND, ND, ND);
  cvt_t_kernel<<<cvg, cvb, 0, stream>>>(w1, w1_t, NL, ND, NF, NF);
  cvt_t_kernel<<<cvg, cvb, 0, stream>>>(w2, w2_t, NL, NF, ND, ND);
  cvt_t_kernel<<<cvg, cvb, 0, stream>>>(fcw, fc_t, 1, ND, NV, NVP);

  embed_kernel<<<dim3(NM), dim3(256), 0, stream>>>(x, tok, pos, h);

  const dim3 gblk(256);
  const dim3 gD(ND / 128, NM / 128);    // (8, 32)
  const dim3 gF(NF / 128, NM / 128);    // (32, 32)
  const dim3 gV(NVP / 128, NM / 128);   // (393, 32)
  const dim3 agrid(NT / 64, NB * NH), ablk(128);

  for (int l = 0; l < NL; ++l) {
    ln_kernel<<<dim3(NM), dim3(256), 0, stream>>>(
        h, ln1g + (size_t)l * ND, ln1b + (size_t)l * ND, hnf, hnb);
    gemm_bf16_kernel<<<gD, gblk, 0, stream>>>(
        hnb, wq_t + (size_t)l * ND * ND, bq + (size_t)l * ND, nullptr, nullptr,
        qb, ND, ND, ND, 0);
    gemm_bf16_kernel<<<gD, gblk, 0, stream>>>(
        hnb, wk_t + (size_t)l * ND * ND, bk + (size_t)l * ND, nullptr, nullptr,
        kb, ND, ND, ND, 0);
    gemm_bf16_kernel<<<gD, gblk, 0, stream>>>(
        hnb, wv_t + (size_t)l * ND * ND, bv + (size_t)l * ND, nullptr, nullptr,
        vb, ND, ND, ND, 0);
    attn_kernel<<<agrid, ablk, 0, stream>>>(qb, kb, vb, ab);
    // h = hn + attn @ wo + bo   (residual around post-norm activation)
    gemm_bf16_kernel<<<gD, gblk, 0, stream>>>(
        ab, wo_t + (size_t)l * ND * ND, bo + (size_t)l * ND, hnf, h, nullptr,
        ND, ND, ND, 0);
    ln_kernel<<<dim3(NM), dim3(256), 0, stream>>>(
        h, ln2g + (size_t)l * ND, ln2b + (size_t)l * ND, hnf, hnb);
    gemm_bf16_kernel<<<gF, gblk, 0, stream>>>(
        hnb, w1_t + (size_t)l * ND * NF, b1 + (size_t)l * NF, nullptr, nullptr,
        ffnb, NF, ND, NF, 1);
    gemm_bf16_kernel<<<gD, gblk, 0, stream>>>(
        ffnb, w2_t + (size_t)l * NF * ND, b2 + (size_t)l * ND, hnf, h, nullptr,
        ND, NF, ND, 0);
  }

  ln_kernel<<<dim3(NM), dim3(256), 0, stream>>>(h, lnfg, lnfb, hnf, hnb);
  gemm_bf16_kernel<<<gV, gblk, 0, stream>>>(hnb, fc_t, fcb, nullptr, logits,
                                            nullptr, NVP, ND, NV, 0);
}
